// TransformerBlock_35467839931007
// MI455X (gfx1250) — matrix operand order repaired and checked by reading, not compiled, not run on hardware
//
#include <hip/hip_runtime.h>
#include <hip/hip_bf16.h>

#define DMODEL 2048
#define DFF    8192
#define HEADS  16
#define DK     128
#define SEQL   2048
#define MS     4096   // BATCH * SEQ rows

typedef __attribute__((ext_vector_type(16))) __bf16 bf16x16;
typedef __attribute__((ext_vector_type(8)))  float  f32x8;

union BF16Frag { uint4 u[2]; bf16x16 v; };

__device__ inline unsigned short f2bf(float f) {
  unsigned int u = __builtin_bit_cast(unsigned int, f);
  u = (u + 0x7FFFu + ((u >> 16) & 1u)) >> 16;   // round-to-nearest-even
  return (unsigned short)u;
}

__device__ inline f32x8 wmma_bf16(bf16x16 a, bf16x16 b, f32x8 c) {
  // D(16x16 f32) = A(16x32 bf16) * B(32x16 bf16) + C
  return __builtin_amdgcn_wmma_f32_16x16x32_bf16(false, a, false, b, (short)0, c,
                                                 false, false);
}

// ---------------------------------------------------------------------------
// fp32 -> bf16 straight convert
// ---------------------------------------------------------------------------
__global__ void cvt_kernel(const float* __restrict__ in,
                           unsigned short* __restrict__ out, size_t n) {
  size_t i = (size_t)blockIdx.x * blockDim.x + threadIdx.x;
  size_t stride = (size_t)gridDim.x * blockDim.x;
  for (; i < n; i += stride) out[i] = f2bf(in[i]);
}

// ---------------------------------------------------------------------------
// W (K x N fp32, row-major) -> Wt (N x K bf16, row-major)  [32x32 LDS tiles]
// ---------------------------------------------------------------------------
__global__ void transpose_cvt_kernel(const float* __restrict__ W,
                                     unsigned short* __restrict__ Wt,
                                     int K, int N) {
  __shared__ float tile[32][33];
  const int n0 = blockIdx.x * 32, k0 = blockIdx.y * 32;
  const int tx = threadIdx.x, ty = threadIdx.y;
#pragma unroll
  for (int i = 0; i < 4; ++i) {
    int kk = ty + i * 8;
    tile[kk][tx] = W[(size_t)(k0 + kk) * N + n0 + tx];
  }
  __syncthreads();
#pragma unroll
  for (int i = 0; i < 4; ++i) {
    int nn = ty + i * 8;
    Wt[(size_t)(n0 + nn) * K + k0 + tx] = f2bf(tile[tx][nn]);
  }
}

// ---------------------------------------------------------------------------
// GEMM: C[M,N] = A[M,K](bf16) * Bt[N,K](bf16, pre-transposed) + bias
//   optional: += resid (fp32 M*N), ReLU, fp32 out, bf16 out
// Block: 256 thr (8 waves). Block tile 128x128, double-buffered LDS.
// Wave tile 32x64 (2x4 WMMA tiles): 8 WMMA per 32-deep k-step per wave.
// ---------------------------------------------------------------------------
__global__ __launch_bounds__(256) void gemm_bf16_kernel(
    const unsigned short* __restrict__ A, const unsigned short* __restrict__ Bt,
    const float* __restrict__ bias, const float* __restrict__ resid,
    float* __restrict__ outF, unsigned short* __restrict__ outB,
    int M, int N, int K, int relu) {
  // stride 40 halfwords (80B rows) -> 16B-aligned b128 frags, conflict-free banks
  __shared__ __align__(16) unsigned short As[2][128 * 40];
  __shared__ __align__(16) unsigned short Bs[2][128 * 40];
  const int t = threadIdx.x;
  const int lane = t & 31, wave = t >> 5;
  const int wm = wave & 3, wn = wave >> 2;
  const int r = lane & 15, hf = lane >> 4;
  const int m0 = blockIdx.y * 128;
  const int n0 = blockIdx.x * 128;

  f32x8 c[2][4] = {};

  auto stage = [&](int k0, int buf) {
    unsigned short* As_ = &As[buf][0];
    unsigned short* Bs_ = &Bs[buf][0];
#pragma unroll
    for (int i = 0; i < 2; ++i) {
      int idx = t + i * 256;
      int row = idx >> 2, q = idx & 3;
      uint4 av = *reinterpret_cast<const uint4*>(
          &A[(size_t)(m0 + row) * K + k0 + q * 8]);
      uint4 bv = *reinterpret_cast<const uint4*>(
          &Bt[(size_t)(n0 + row) * K + k0 + q * 8]);
      *reinterpret_cast<uint4*>(&As_[row * 40 + q * 8]) = av;
      *reinterpret_cast<uint4*>(&Bs_[row * 40 + q * 8]) = bv;
    }
  };

  stage(0, 0);
  __syncthreads();
  const int nk = K >> 5;
  for (int kt = 0; kt < nk; ++kt) {
    const int buf = kt & 1;
    if (kt + 1 < nk) stage((kt + 1) << 5, buf ^ 1);  // overlap next-tile staging

    const unsigned short* As_ = &As[buf][0];
    const unsigned short* Bs_ = &Bs[buf][0];
    BF16Frag a[2], b[4];
#pragma unroll
    for (int i = 0; i < 2; ++i) {  // A layout: K = hf*8+{0..7} and 16+hf*8+{0..7}
      int rb = (wm * 32 + i * 16 + r) * 40;
      a[i].u[0] = *reinterpret_cast<const uint4*>(&As_[rb + hf * 8]);
      a[i].u[1] = *reinterpret_cast<const uint4*>(&As_[rb + 16 + hf * 8]);
    }
#pragma unroll
    for (int j = 0; j < 4; ++j) {  // B layout: K = hf*8+{0..7} and 16+hf*8+{0..7}
      int cb = (wn * 64 + j * 16 + r) * 40;
      b[j].u[0] = *reinterpret_cast<const uint4*>(&Bs_[cb + hf * 8]);
      b[j].u[1] = *reinterpret_cast<const uint4*>(&Bs_[cb + 16 + hf * 8]);
    }
#pragma unroll
    for (int i = 0; i < 2; ++i)
#pragma unroll
      for (int j = 0; j < 4; ++j) c[i][j] = wmma_bf16(a[i].v, b[j].v, c[i][j]);
    __syncthreads();
  }

  // Epilogue: C layout row = v + 8*hf, col = lane&15
#pragma unroll
  for (int i = 0; i < 2; ++i) {
#pragma unroll
    for (int j = 0; j < 4; ++j) {
      int col = n0 + wn * 64 + j * 16 + r;
      float bval = bias ? bias[col] : 0.0f;
#pragma unroll
      for (int v = 0; v < 8; ++v) {
        int row = m0 + wm * 32 + i * 16 + v + 8 * hf;
        float val = c[i][j][v] + bval;
        if (resid) val += resid[(size_t)row * N + col];
        if (relu) val = fmaxf(val, 0.0f);
        if (outF) outF[(size_t)row * N + col] = val;
        if (outB) outB[(size_t)row * N + col] = f2bf(val);
      }
    }
  }
}

// ---------------------------------------------------------------------------
// Flash attention: one block = 128 query rows of one (batch, head).
// 8 waves, each owns 16 q-rows; K/V tiles (32 keys) shared via LDS.
// ---------------------------------------------------------------------------
__global__ __launch_bounds__(256) void attn_kernel(
    const unsigned short* __restrict__ Qb, const unsigned short* __restrict__ Kb,
    const unsigned short* __restrict__ Vb, unsigned short* __restrict__ Ob) {
  __shared__ __align__(16) unsigned short Kt[32 * 136];   // [key][d], 272B rows
  __shared__ __align__(16) unsigned short Vt[128 * 40];   // transposed [d][key]
  __shared__ __align__(16) unsigned short Pt[8 * 16 * 32];// per-wave P staging
  const int t = threadIdx.x, lane = t & 31, wave = t >> 5;
  const int r = lane & 15, hf = lane >> 4;
  const int b = blockIdx.x / HEADS, h = blockIdx.x % HEADS;
  const int q0 = blockIdx.y * 128;
  const size_t rowbase = (size_t)b * SEQL;
  const int hcol = h * DK;

  // Q fragments (16 rows x 128, 4 k-steps) held in registers for the whole pass
  BF16Frag qf[4];
  {
    const unsigned short* qp =
        Qb + (rowbase + q0 + wave * 16 + r) * DMODEL + hcol;
#pragma unroll
    for (int ks = 0; ks < 4; ++ks) {
      qf[ks].u[0] = *reinterpret_cast<const uint4*>(&qp[ks * 32 + hf * 8]);
      qf[ks].u[1] = *reinterpret_cast<const uint4*>(&qp[ks * 32 + 16 + hf * 8]);
    }
  }

  f32x8 o[8] = {};
  f32x8 mrow, lrow;
#pragma unroll
  for (int v = 0; v < 8; ++v) { mrow[v] = -3.0e38f; lrow[v] = 0.0f; }
  const float scale = 0.08838834764831845f * 1.4426950408889634f; // rsqrt(128)*log2(e)
  const int jtiles = q0 / 32 + 4;  // causal extent for this 128-row block

  for (int j = 0; j < jtiles; ++j) {
    const int kbase = j * 32;
    // Stage K tile (row-major) and V tile (transposed) — 2 uint4 per thread each
#pragma unroll
    for (int i = 0; i < 2; ++i) {
      int idx = t + i * 256;
      int krow = idx >> 4, q = idx & 15;
      uint4 src = *reinterpret_cast<const uint4*>(
          &Kb[(rowbase + kbase + krow) * DMODEL + hcol + q * 8]);
      *reinterpret_cast<uint4*>(&Kt[krow * 136 + q * 8]) = src;
    }
#pragma unroll
    for (int i = 0; i < 2; ++i) {
      int idx = t + i * 256;
      int krow = idx >> 4, q = idx & 15;
      uint4 src = *reinterpret_cast<const uint4*>(
          &Vb[(rowbase + kbase + krow) * DMODEL + hcol + q * 8]);
      const unsigned short* s16 = reinterpret_cast<const unsigned short*>(&src);
#pragma unroll
      for (int e = 0; e < 8; ++e) Vt[(q * 8 + e) * 40 + krow] = s16[e];
    }
    __syncthreads();

    // S block (16 q x 32 keys) = Q * K^T : two N-tiles x four k-steps
    f32x8 s[2] = {};
#pragma unroll
    for (int nt = 0; nt < 2; ++nt) {
#pragma unroll
      for (int ks = 0; ks < 4; ++ks) {
        BF16Frag kb;
        int cb = (nt * 16 + r) * 136 + ks * 32;
        kb.u[0] = *reinterpret_cast<const uint4*>(&Kt[cb + hf * 8]);
        kb.u[1] = *reinterpret_cast<const uint4*>(&Kt[cb + 16 + hf * 8]);
        s[nt] = wmma_bf16(qf[ks].v, kb.v, s[nt]);
      }
    }

    // scale + causal mask + online softmax (row = v + 8*hf in C layout)
    f32x8 alpha;
#pragma unroll
    for (int v = 0; v < 8; ++v) {
      int rowg = q0 + wave * 16 + v + 8 * hf;
      float s0 = (kbase + r      <= rowg) ? s[0][v] * scale : -3.0e38f;
      float s1 = (kbase + 16 + r <= rowg) ? s[1][v] * scale : -3.0e38f;
      float mx = fmaxf(s0, s1);
      mx = fmaxf(mx, __shfl_xor(mx, 1));
      mx = fmaxf(mx, __shfl_xor(mx, 2));
      mx = fmaxf(mx, __shfl_xor(mx, 4));
      mx = fmaxf(mx, __shfl_xor(mx, 8));
      float mn = fmaxf(mrow[v], mx);
      alpha[v] = exp2f(mrow[v] - mn);
      mrow[v] = mn;
      float p0 = exp2f(s0 - mn);
      float p1 = exp2f(s1 - mn);
      s[0][v] = p0; s[1][v] = p1;
      float sum = p0 + p1;
      sum += __shfl_xor(sum, 1);
      sum += __shfl_xor(sum, 2);
      sum += __shfl_xor(sum, 4);
      sum += __shfl_xor(sum, 8);
      lrow[v] = lrow[v] * alpha[v] + sum;
    }

    // C-layout -> A-layout via wave-private LDS patch (bf16)
    unsigned short* pw = &Pt[wave * 512];
#pragma unroll
    for (int nt = 0; nt < 2; ++nt)
#pragma unroll
      for (int v = 0; v < 8; ++v)
        pw[(v + 8 * hf) * 32 + nt * 16 + r] = f2bf(s[nt][v]);
    asm volatile("s_wait_dscnt 0x0" ::: "memory");  // wave-local LDS RAW fence
    BF16Frag pa;
    pa.u[0] = *reinterpret_cast<const uint4*>(&pw[r * 32 + hf * 8]);
    pa.u[1] = *reinterpret_cast<const uint4*>(&pw[r * 32 + 16 + hf * 8]);

    // O = O*alpha + P*V  (8 output N-tiles over d_k=128)
#pragma unroll
    for (int nt = 0; nt < 8; ++nt) {
#pragma unroll
      for (int v = 0; v < 8; ++v) o[nt][v] *= alpha[v];
      BF16Frag vb;
      int cb = (nt * 16 + r) * 40;
      vb.u[0] = *reinterpret_cast<const uint4*>(&Vt[cb + hf * 8]);
      vb.u[1] = *reinterpret_cast<const uint4*>(&Vt[cb + 16 + hf * 8]);
      o[nt] = wmma_bf16(pa.v, vb.v, o[nt]);
    }
    __syncthreads();
  }

#pragma unroll
  for (int v = 0; v < 8; ++v) lrow[v] = 1.0f / lrow[v];
#pragma unroll
  for (int nt = 0; nt < 8; ++nt)
#pragma unroll
    for (int v = 0; v < 8; ++v) {
      int rowg = q0 + wave * 16 + v + 8 * hf;
      Ob[(rowbase + rowg) * DMODEL + hcol + nt * 16 + r] =
          f2bf(o[nt][v] * lrow[v]);
    }
}

// ---------------------------------------------------------------------------
// Row LayerNorm over DMODEL; writes fp32 and optional bf16
// ---------------------------------------------------------------------------
__global__ __launch_bounds__(256) void layernorm_kernel(
    const float* __restrict__ in, const float* __restrict__ g,
    const float* __restrict__ be, float* __restrict__ outF,
    unsigned short* __restrict__ outB) {
  __shared__ float red[256], red2[256];
  const int row = blockIdx.x;
  const float* x = in + (size_t)row * DMODEL;
  float s = 0.0f, s2 = 0.0f;
  for (int i = threadIdx.x; i < DMODEL; i += 256) {
    float v = x[i]; s += v; s2 += v * v;
  }
  red[threadIdx.x] = s; red2[threadIdx.x] = s2;
  __syncthreads();
  for (int st = 128; st > 0; st >>= 1) {
    if (threadIdx.x < st) {
      red[threadIdx.x] += red[threadIdx.x + st];
      red2[threadIdx.x] += red2[threadIdx.x + st];
    }
    __syncthreads();
  }
  const float mu = red[0] * (1.0f / DMODEL);
  const float var = red2[0] * (1.0f / DMODEL) - mu * mu;
  const float rs = rsqrtf(var + 1e-5f);
  for (int i = threadIdx.x; i < DMODEL; i += 256) {
    float y = (x[i] - mu) * rs * g[i] + be[i];
    if (outF) outF[(size_t)row * DMODEL + i] = y;
    if (outB) outB[(size_t)row * DMODEL + i] = f2bf(y);
  }
}

// ---------------------------------------------------------------------------
extern "C" void kernel_launch(void* const* d_in, const int* in_sizes, int n_in,
                              void* d_out, int out_size, void* d_ws,
                              size_t ws_size, hipStream_t stream) {
  const float* x   = (const float*)d_in[0];
  // d_in[1] is the tril mask: causal, applied analytically in attn_kernel
  const float* Wq  = (const float*)d_in[2];
  const float* bq  = (const float*)d_in[3];
  const float* Wk  = (const float*)d_in[4];
  const float* bk  = (const float*)d_in[5];
  const float* Wv  = (const float*)d_in[6];
  const float* bv  = (const float*)d_in[7];
  const float* Wo  = (const float*)d_in[8];
  const float* bo  = (const float*)d_in[9];
  const float* g1  = (const float*)d_in[10];
  const float* be1 = (const float*)d_in[11];
  const float* W1  = (const float*)d_in[12];
  const float* b1  = (const float*)d_in[13];
  const float* W2  = (const float*)d_in[14];
  const float* b2  = (const float*)d_in[15];
  const float* g2  = (const float*)d_in[16];
  const float* be2 = (const float*)d_in[17];
  float* out = (float*)d_out;

  char* ws = (char*)d_ws;
  size_t off = 0;
  auto alloc = [&](size_t bytes) -> void* {
    void* p = ws + off;
    off += (bytes + 255) & ~(size_t)255;
    return p;
  };
  unsigned short* xb   = (unsigned short*)alloc((size_t)MS * DMODEL * 2);
  unsigned short* WqT  = (unsigned short*)alloc((size_t)DMODEL * DMODEL * 2);
  unsigned short* WkT  = (unsigned short*)alloc((size_t)DMODEL * DMODEL * 2);
  unsigned short* WvT  = (unsigned short*)alloc((size_t)DMODEL * DMODEL * 2);
  unsigned short* WoT  = (unsigned short*)alloc((size_t)DMODEL * DMODEL * 2);
  unsigned short* W1T  = (unsigned short*)alloc((size_t)DFF * DMODEL * 2);
  unsigned short* W2T  = (unsigned short*)alloc((size_t)DMODEL * DFF * 2);
  unsigned short* Qb   = (unsigned short*)alloc((size_t)MS * DMODEL * 2);
  unsigned short* Kb   = (unsigned short*)alloc((size_t)MS * DMODEL * 2);
  unsigned short* Vb   = (unsigned short*)alloc((size_t)MS * DMODEL * 2);
  unsigned short* ctxb = (unsigned short*)alloc((size_t)MS * DMODEL * 2);
  float*          res1 = (float*)alloc((size_t)MS * DMODEL * 4);
  float*          hbuf = (float*)alloc((size_t)MS * DMODEL * 4);
  unsigned short* hb   = (unsigned short*)alloc((size_t)MS * DMODEL * 2);
  unsigned short* f1b  = (unsigned short*)alloc((size_t)MS * DFF * 2);
  float*          res2 = (float*)alloc((size_t)MS * DMODEL * 4);

  // 1) precision conversion (x straight; weights transposed to N-major)
  cvt_kernel<<<2048, 256, 0, stream>>>(x, xb, (size_t)MS * DMODEL);
  dim3 tb(32, 8);
  transpose_cvt_kernel<<<dim3(64, 64), tb, 0, stream>>>(Wq, WqT, DMODEL, DMODEL);
  transpose_cvt_kernel<<<dim3(64, 64), tb, 0, stream>>>(Wk, WkT, DMODEL, DMODEL);
  transpose_cvt_kernel<<<dim3(64, 64), tb, 0, stream>>>(Wv, WvT, DMODEL, DMODEL);
  transpose_cvt_kernel<<<dim3(64, 64), tb, 0, stream>>>(Wo, WoT, DMODEL, DMODEL);
  transpose_cvt_kernel<<<dim3(256, 64), tb, 0, stream>>>(W1, W1T, DMODEL, DFF);
  transpose_cvt_kernel<<<dim3(64, 256), tb, 0, stream>>>(W2, W2T, DFF, DMODEL);

  // 2) Q/K/V projections (bf16 outputs for attention)
  dim3 gD(DMODEL / 128, MS / 128);
  gemm_bf16_kernel<<<gD, 256, 0, stream>>>(xb, WqT, bq, nullptr, nullptr, Qb,
                                           MS, DMODEL, DMODEL, 0);
  gemm_bf16_kernel<<<gD, 256, 0, stream>>>(xb, WkT, bk, nullptr, nullptr, Kb,
                                           MS, DMODEL, DMODEL, 0);
  gemm_bf16_kernel<<<gD, 256, 0, stream>>>(xb, WvT, bv, nullptr, nullptr, Vb,
                                           MS, DMODEL, DMODEL, 0);

  // 3) causal flash attention -> ctx (bf16)
  attn_kernel<<<dim3(2 * HEADS, SEQL / 128), 256, 0, stream>>>(Qb, Kb, Vb, ctxb);

  // 4) output projection + residual(x) -> res1 (fp32)
  gemm_bf16_kernel<<<gD, 256, 0, stream>>>(ctxb, WoT, bo, x, res1, nullptr,
                                           MS, DMODEL, DMODEL, 0);
  // 5) LN1 -> h (fp32 + bf16)
  layernorm_kernel<<<MS, 256, 0, stream>>>(res1, g1, be1, hbuf, hb);

  // 6) FFN
  gemm_bf16_kernel<<<dim3(DFF / 128, MS / 128), 256, 0, stream>>>(
      hb, W1T, b1, nullptr, nullptr, f1b, MS, DFF, DMODEL, 1);
  gemm_bf16_kernel<<<gD, 256, 0, stream>>>(f1b, W2T, b2, hbuf, res2, nullptr,
                                           MS, DMODEL, DFF, 0);
  // 7) LN2 -> output
  layernorm_kernel<<<MS, 256, 0, stream>>>(res2, g2, be2, out, nullptr);
}
